// NN_45260365365928
// MI455X (gfx1250) — matrix operand order repaired and checked by reading, not compiled, not run on hardware
//
#include <hip/hip_runtime.h>
#include <hip/hip_bf16.h>
#include <math.h>

typedef __attribute__((ext_vector_type(16))) __bf16 v16bf;
typedef __attribute__((ext_vector_type(8)))  float  v8f;

struct alignas(16) U4 { unsigned x, y, z, w; };
union Frag { U4 u[2]; v16bf v; };

#define HDIM   256
#define BATCH  1024
#define EDIM   300
#define SEGLEN 200
#define DIN    903
#define KPAD   928
#define G4H    1024
#define EPSV   1e-8f

// ---------------- prep kernels ----------------
__global__ void k_cvt_pad(const float* __restrict__ src, __bf16* __restrict__ dst,
                          int K, int Kpad, int total) {
  int i = blockIdx.x * blockDim.x + threadIdx.x;
  if (i >= total) return;
  int r = i / Kpad, k = i - r * Kpad;
  dst[i] = (k < K) ? (__bf16)src[r * K + k] : (__bf16)0.0f;
}

__global__ void k_transpose(const float* __restrict__ w, float* __restrict__ wT) {
  int i = blockIdx.x * blockDim.x + threadIdx.x;   // over 1024*256
  if (i >= G4H * HDIM) return;
  int n = i / HDIM, k = i - n * HDIM;
  wT[k * G4H + n] = w[i];
}

__global__ void k_bias(const float* __restrict__ a, const float* __restrict__ b,
                       float* __restrict__ o) {
  int i = blockIdx.x * blockDim.x + threadIdx.x;
  if (i < G4H) o[i] = a[i] + b[i];
}

// ---------------- conv: embedding means + cosines -> bf16 xs ----------------
__device__ __forceinline__ float block_reduce256(float v, float* red) {
  int tid = threadIdx.x;
  red[tid] = v; __syncthreads();
  #pragma unroll
  for (int s = 128; s > 0; s >>= 1) {
    if (tid < s) red[tid] += red[tid + s];
    __syncthreads();
  }
  float r = red[0]; __syncthreads();
  return r;
}

__global__ __launch_bounds__(256)
void k_conv(const int* __restrict__ x, const float* __restrict__ emb,
            __bf16* __restrict__ xs) {
  const int b = blockIdx.x, tid = threadIdx.x;
  __shared__ int   xrow[600];
  __shared__ float seg[3][EDIM];
  __shared__ float red[256];
  for (int i = tid; i < 600; i += 256) xrow[i] = x[b * 600 + i];
  __syncthreads();
  for (int s = 0; s < 3; ++s) {
    float a0 = 0.f, a1 = 0.f; int cnt = 0;
    for (int t = 0; t < SEGLEN; ++t) {
      int idx = xrow[s * SEGLEN + t];
      cnt += (idx != 0);
      const float* e = emb + (size_t)idx * EDIM;
      a0 += e[tid];
      if (tid < EDIM - 256) a1 += e[256 + tid];
    }
    float inv = 1.0f / (float)cnt;
    seg[s][tid] = a0 * inv;
    if (tid < EDIM - 256) seg[s][256 + tid] = a1 * inv;
  }
  __syncthreads();
  float t0 = seg[0][tid], u0 = seg[1][tid], j0 = seg[2][tid];
  float t1 = 0.f, u1 = 0.f, j1 = 0.f;
  if (tid < EDIM - 256) { t1 = seg[0][256+tid]; u1 = seg[1][256+tid]; j1 = seg[2][256+tid]; }
  float ntt = block_reduce256(t0*t0 + t1*t1, red);
  float nuu = block_reduce256(u0*u0 + u1*u1, red);
  float njj = block_reduce256(j0*j0 + j1*j1, red);
  float dtu = block_reduce256(t0*u0 + t1*u1, red);
  float duj = block_reduce256(u0*j0 + u1*j1, red);
  float dtj = block_reduce256(t0*j0 + t1*j1, red);
  float nt = fmaxf(sqrtf(ntt), EPSV), nu = fmaxf(sqrtf(nuu), EPSV), nj = fmaxf(sqrtf(njj), EPSV);
  __bf16* row = xs + (size_t)b * KPAD;
  row[tid]       = (__bf16)t0;
  row[301 + tid] = (__bf16)u0;
  row[602 + tid] = (__bf16)j0;
  if (tid < EDIM - 256) {
    row[256 + tid] = (__bf16)t1;   // st dims 256..299
    row[557 + tid] = (__bf16)u1;   // su dims 256..299 at cols 557..600
    row[858 + tid] = (__bf16)j1;   // ju dims 256..299 at cols 858..901
  }
  if (tid == 0) {
    row[300] = (__bf16)(dtu / (nt * nu));
    row[601] = (__bf16)(duj / (nu * nj));
    row[902] = (__bf16)(dtj / (nt * nj));
  }
  if (tid < KPAD - DIN) row[DIN + tid] = (__bf16)0.0f;  // zero pad cols 903..927
}

// ---------------- WMMA GEMM: C(1024x1024) = A(1024xK) * Bw(1024xK)^T + bias[n] ----------------
__global__ __launch_bounds__(256)
void k_wmma_gemm(const __bf16* __restrict__ A, const __bf16* __restrict__ Bw,
                 const float* __restrict__ bias, float* __restrict__ C, int K) {
  const int tid   = threadIdx.x;
  const int lane  = tid & 31;
  const int w     = tid >> 5;        // 0..7
  const int wr    = w >> 2;          // 0..1 -> 64 rows each
  const int wc    = w & 3;           // 0..3 -> 32 cols each
  const int rowbase = blockIdx.x * 128 + wr * 64;
  const int colbase = blockIdx.y * 128 + wc * 32;
  const int lhalf = (lane >= 16) ? 1 : 0;
  const int l16   = lane & 15;

  v8f acc[4][2];
  #pragma unroll
  for (int j = 0; j < 2; ++j) {
    float bv = bias[colbase + j * 16 + l16];
    #pragma unroll
    for (int i = 0; i < 4; ++i)
      #pragma unroll
      for (int r = 0; r < 8; ++r) acc[i][j][r] = bv;
  }

  for (int kb = 0; kb < K; kb += 32) {
    Frag a[4], bf[2];
    #pragma unroll
    for (int i = 0; i < 4; ++i) {
      // A 16x32 bf16 layout: lane holds row l16, K = {base..base+7, base+16..base+23}, base = lhalf*8
      const __bf16* p = A + (size_t)(rowbase + i * 16 + l16) * K + kb + (lhalf ? 8 : 0);
      a[i].u[0] = *(const U4*)p;
      a[i].u[1] = *(const U4*)(p + 16);
    }
    #pragma unroll
    for (int j = 0; j < 2; ++j) {
      // B 32x16 bf16 layout: lane holds column l16, K = {base..base+7, base+16..base+23}, base = lhalf*8
      const __bf16* p = Bw + (size_t)(colbase + j * 16 + l16) * K + kb + (lhalf ? 8 : 0);
      bf[j].u[0] = *(const U4*)p;
      bf[j].u[1] = *(const U4*)(p + 16);
    }
    #pragma unroll
    for (int i = 0; i < 4; ++i)
      #pragma unroll
      for (int j = 0; j < 2; ++j)
        acc[i][j] = __builtin_amdgcn_wmma_f32_16x16x32_bf16(
            false, a[i].v, false, bf[j].v, (short)0, acc[i][j], false, false);
  }

  #pragma unroll
  for (int i = 0; i < 4; ++i)
    #pragma unroll
    for (int j = 0; j < 2; ++j)
      #pragma unroll
      for (int r = 0; r < 8; ++r) {
        int m = rowbase + i * 16 + r + (lhalf ? 8 : 0);
        int n = colbase + j * 16 + l16;
        C[(size_t)m * G4H + n] = acc[i][j][r];
      }
}

// ---------------- sequential LSTM recurrence (one WG, h in LDS, c in registers) ----------------
__global__ __launch_bounds__(1024)
void k_lstm(const float* __restrict__ G, const float* __restrict__ whhT,
            const float* __restrict__ h0, const float* __restrict__ c0,
            __bf16* __restrict__ ys_bf,
            const float* __restrict__ fcw, const float* __restrict__ fcb,
            float* __restrict__ sig_out,
            float* __restrict__ h_fin, float* __restrict__ c_fin) {
  const int tid = threadIdx.x;
  __shared__ float h_s[HDIM];
  __shared__ float g_s[G4H];
  __shared__ float red[HDIM];
  float c_reg = 0.f;
  if (tid < HDIM) { h_s[tid] = h0[tid]; c_reg = c0[tid]; }
  __syncthreads();
  for (int t = 0; t < BATCH; ++t) {
    float acc = G[(size_t)t * G4H + tid];     // precomputed x@W_ih^T + biases
    #pragma unroll 8
    for (int k = 0; k < HDIM; ++k) acc += h_s[k] * whhT[k * G4H + tid];  // coalesced
    g_s[tid] = acc;
    __syncthreads();
    if (tid < HDIM) {
      float ig = 1.f / (1.f + expf(-g_s[tid]));
      float fg = 1.f / (1.f + expf(-g_s[tid + HDIM]));
      float gg = tanhf(g_s[tid + 2 * HDIM]);
      float og = 1.f / (1.f + expf(-g_s[tid + 3 * HDIM]));
      c_reg = fg * c_reg + ig * gg;
      float hv = og * tanhf(c_reg);
      h_s[tid] = hv;
      if (ys_bf)   ys_bf[(size_t)t * HDIM + tid] = (__bf16)hv;
      if (sig_out) red[tid] = hv * fcw[tid];
    }
    __syncthreads();
    if (sig_out && tid < 32) {
      float p = 0.f;
      #pragma unroll
      for (int m = 0; m < 8; ++m) p += red[tid * 8 + m];
      #pragma unroll
      for (int off = 16; off > 0; off >>= 1) p += __shfl_down(p, off, 32);
      if (tid == 0) sig_out[t] = 1.f / (1.f + expf(-(p + fcb[0])));
    }
  }
  if (tid < HDIM) { h_fin[tid] = h_s[tid]; c_fin[tid] = c_reg; }
}

// ---------------- launch ----------------
extern "C" void kernel_launch(void* const* d_in, const int* in_sizes, int n_in,
                              void* d_out, int out_size, void* d_ws, size_t ws_size,
                              hipStream_t stream) {
  (void)in_sizes; (void)n_in; (void)out_size; (void)ws_size;
  const int*   x     = (const int*)d_in[0];
  const float* h0    = (const float*)d_in[1];
  const float* c0    = (const float*)d_in[2];
  const float* emb   = (const float*)d_in[3];
  const float* w_ih0 = (const float*)d_in[4];
  const float* w_hh0 = (const float*)d_in[5];
  const float* b_ih0 = (const float*)d_in[6];
  const float* b_hh0 = (const float*)d_in[7];
  const float* w_ih1 = (const float*)d_in[8];
  const float* w_hh1 = (const float*)d_in[9];
  const float* b_ih1 = (const float*)d_in[10];
  const float* b_hh1 = (const float*)d_in[11];
  const float* fc_w  = (const float*)d_in[12];
  const float* fc_b  = (const float*)d_in[13];
  float* out = (float*)d_out;    // [sig(1024) | hn(2x256) | cn(2x256)]

  char* ws = (char*)d_ws;
  size_t off = 0;
  auto take = [&](size_t bytes) { char* p = ws + off; off += (bytes + 255) & ~(size_t)255; return p; };
  __bf16* xs_bf   = (__bf16*)take((size_t)BATCH * KPAD * 2);
  __bf16* wih0_bf = (__bf16*)take((size_t)G4H  * KPAD * 2);
  __bf16* wih1_bf = (__bf16*)take((size_t)G4H  * HDIM * 2);
  float*  whh0T   = (float*) take((size_t)HDIM * G4H * 4);
  float*  whh1T   = (float*) take((size_t)HDIM * G4H * 4);
  float*  bias0   = (float*) take((size_t)G4H * 4);
  float*  bias1   = (float*) take((size_t)G4H * 4);
  float*  G0      = (float*) take((size_t)BATCH * G4H * 4);
  float*  G1      = (float*) take((size_t)BATCH * G4H * 4);
  __bf16* ys1_bf  = (__bf16*)take((size_t)BATCH * HDIM * 2);

  { int total = G4H * KPAD; k_cvt_pad<<<(total + 255) / 256, 256, 0, stream>>>(w_ih0, wih0_bf, DIN, KPAD, total); }
  { int total = G4H * HDIM; k_cvt_pad<<<(total + 255) / 256, 256, 0, stream>>>(w_ih1, wih1_bf, HDIM, HDIM, total); }
  k_transpose<<<(G4H * HDIM + 255) / 256, 256, 0, stream>>>(w_hh0, whh0T);
  k_transpose<<<(G4H * HDIM + 255) / 256, 256, 0, stream>>>(w_hh1, whh1T);
  k_bias<<<4, 256, 0, stream>>>(b_ih0, b_hh0, bias0);
  k_bias<<<4, 256, 0, stream>>>(b_ih1, b_hh1, bias1);

  k_conv<<<BATCH, 256, 0, stream>>>(x, emb, xs_bf);

  k_wmma_gemm<<<dim3(8, 8), 256, 0, stream>>>(xs_bf, wih0_bf, bias0, G0, KPAD);
  k_lstm<<<1, 1024, 0, stream>>>(G0, whh0T, h0, c0, ys1_bf,
                                 nullptr, nullptr, nullptr,
                                 out + 1024, out + 1536);
  k_wmma_gemm<<<dim3(8, 8), 256, 0, stream>>>(ys1_bf, wih1_bf, bias1, G1, HDIM);
  k_lstm<<<1, 1024, 0, stream>>>(G1, whh1T, h0 + HDIM, c0 + HDIM, nullptr,
                                 fc_w, fc_b, out,
                                 out + 1024 + HDIM, out + 1536 + HDIM);
}
